// ProgressiveWaveContextEncoder_37933151159073
// MI455X (gfx1250) — matrix operand order repaired and checked by reading, not compiled, not run on hardware
//
#include <hip/hip_runtime.h>
#include <hip/hip_bf16.h>

// ProgressiveWaveContextEncoder for MI455X (gfx1250, wave32, WMMA).
//
// Key insight: all 10 contexts share prefixes -> ONE 480-step LSTM pass with
// snapshots at t = 263, 287, ..., 479 (vs ~3960 reference steps).
// Per step: gates[16x256] = [h|x|0]{16x96} @ B'{96x256} + bias, done with
// v_wmma_f32_16x16x32_f16 (3 K-chunks x 4 gate tiles per wave).
// Round 2: activation math rewritten to avoid IEEE division fixup chains —
// native v_tanh_f32 if the builtin exists, else v_rcp_f32 fast reciprocal.

typedef _Float16 v16h __attribute__((ext_vector_type(16)));
typedef _Float16 h8   __attribute__((ext_vector_type(8)));
typedef float    v8f  __attribute__((ext_vector_type(8)));

#define HID     64
#define TIN     240
#define TTOT    480
#define MROWS   16      // batch rows per block (WMMA M)
#define THREADS 128     // 4 wave32 waves
#define KPAD    96      // 64 (h) + 3 (x) + 29 (zero pad) -> 3 WMMA k-chunks
#define NCTX    10
#define CSTEP   24

#if defined(__has_builtin)
#if __has_builtin(__builtin_amdgcn_tanhf)
#define USE_NATIVE_TANH 1
#endif
#endif

__device__ __forceinline__ float fast_tanh(float x) {
#ifdef USE_NATIVE_TANH
    return __builtin_amdgcn_tanhf(x);              // v_tanh_f32 (CDNA5 trans op)
#else
    float e2 = __expf(2.0f * x);                   // v_mul + v_exp_f32
    return 1.0f - 2.0f * __builtin_amdgcn_rcpf(e2 + 1.0f);   // v_rcp_f32, no div fixup
#endif
}
__device__ __forceinline__ float fast_sigmoid(float x) {
#ifdef USE_NATIVE_TANH
    return fmaf(0.5f, __builtin_amdgcn_tanhf(0.5f * x), 0.5f);
#else
    return __builtin_amdgcn_rcpf(1.0f + __expf(-x));
#endif
}

__global__ __launch_bounds__(THREADS)
void pwce_lstm_kernel(const float* __restrict__ wi,      // [B,240,3]
                      const float* __restrict__ wfut,    // [B,240,3]
                      const float* __restrict__ W_ih,    // [256,3]
                      const float* __restrict__ W_hh,    // [256,64]
                      const float* __restrict__ b_ih,    // [256]
                      const float* __restrict__ b_hh,    // [256]
                      const float* __restrict__ W_proj,  // [64,64]
                      const float* __restrict__ b_proj,  // [64]
                      float* __restrict__ out)           // [B,10,64]
{
    // x staged once as f16 (one-time global traffic; loop runs out of LDS).
    __shared__ __align__(16) _Float16 xstage[TTOT][MROWS][3];   // 45 KB
    // Double-buffered WMMA A-source rows: [h(64) | x_t(3) | zeros(29)].
    __shared__ __align__(16) _Float16 abuf[2][MROWS][KPAD];     // 6 KB

    const int tid   = threadIdx.x;
    const int w     = tid >> 5;          // wave 0..3 -> hidden slice [16w,16w+16)
    const int l     = tid & 31;          // lane in wave32
    const int nlo   = l & 15;
    const int hi    = l >> 4;            // lane half (ISA fragment layouts)
    const int bbase = blockIdx.x * MROWS;

    // ---- stage x = concat(wave_input, wave_future) along time, f32->f16 ----
    for (int idx = tid; idx < TTOT * MROWS * 3; idx += THREADS) {
        int t = idx / (MROWS * 3);
        int r = idx - t * (MROWS * 3);
        int m = r / 3, d = r - m * 3;
        float v = (t < TIN) ? wi[((size_t)(bbase + m) * TIN + t) * 3 + d]
                            : wfut[((size_t)(bbase + m) * TIN + (t - TIN)) * 3 + d];
        xstage[t][m][d] = (_Float16)v;
    }
    // zero both A-buffers (h_{-1}=0; pad region stays 0 forever)
    for (int idx = tid; idx < 2 * MROWS * KPAD; idx += THREADS)
        ((_Float16*)abuf)[idx] = (_Float16)0.0f;

    // ---- build B' fragments in registers: B'[k][n] ----
    //   k<64  : W_hh[n][k]     (h @ W_hh^T)
    //   64..66: W_ih[n][k-64]  (x @ W_ih^T)
    //   else  : 0
    // B-frag layout (16-bit, 32x16 per chunk): elem j <-> K = 32c + 8*hi + (j&7) + 16*(j>>3), N = nlo.
    v16h bfrag[4][3];
    float gbias[4];
    #pragma unroll
    for (int q = 0; q < 4; ++q) {                 // gate order i,f,g,o
        int n = q * 64 + w * 16 + nlo;
        gbias[q] = b_ih[n] + b_hh[n];
        #pragma unroll
        for (int c = 0; c < 3; ++c) {
            #pragma unroll
            for (int j = 0; j < 16; ++j) {
                int K = 32 * c + hi * 8 + (j & 7) + ((j >> 3) << 4);
                float v = 0.0f;
                if (K < 64)      v = W_hh[n * 64 + K];
                else if (K < 67) v = W_ih[n * 3 + (K - 64)];
                bfrag[q][c][j] = (_Float16)v;
            }
        }
    }
    // projection fragments (K=64 -> 2 chunks), output dim np = 16w + nlo
    v16h pfrag[2];
    const int np = w * 16 + nlo;
    #pragma unroll
    for (int c = 0; c < 2; ++c)
        #pragma unroll
        for (int j = 0; j < 16; ++j)
            pfrag[c][j] = (_Float16)W_proj[np * 64 + (32 * c + hi * 8 + (j & 7) + ((j >> 3) << 4))];
    const float pbias = b_proj[np];

    // per-lane c-state: (m = v + 8*hi, hidden j = np), matches C/D frag layout
    v8f cst;
    #pragma unroll
    for (int v = 0; v < 8; ++v) cst[v] = 0.0f;

    const int arow = nlo;       // A-frag: M = lane&15 for all lanes
    const int koff = hi * 8;    // A-frag lane-half K offset

    __syncthreads();

    int p = 0;
    v16h afr[3];
    for (int t = 0; t < TTOT; ++t) {
        // (a) drop x_t into abuf[p] x-slots (h_{t-1} already there; disjoint k)
        if (tid < MROWS * 3) {
            int m = tid / 3, d = tid - m * 3;
            abuf[p][m][64 + d] = xstage[t][m][d];
        }
        __syncthreads();

        // (c) load A fragments: per ISA, j<8 -> K=32c+koff+j, j>=8 -> +16
        #pragma unroll
        for (int c = 0; c < 3; ++c) {
            const _Float16* base = &abuf[p][arow][32 * c + koff];
            union { v16h v; h8 h[2]; } u;
            u.h[0] = *(const h8*)(base);        // ds_load_b128
            u.h[1] = *(const h8*)(base + 16);   // ds_load_b128
            afr[c] = u.v;
        }

        // snapshot: afr holds h_{t-1}; contexts s=0..8 end at t-1 = 263+24s
        if (t >= 264 && ((t - 264) % CSTEP) == 0) {
            int s = (t - 264) / CSTEP;
            v8f acc;
            #pragma unroll
            for (int v = 0; v < 8; ++v) acc[v] = pbias;
            acc = __builtin_amdgcn_wmma_f32_16x16x32_f16(false, afr[0], false, pfrag[0],
                                                         (short)0, acc, false, false);
            acc = __builtin_amdgcn_wmma_f32_16x16x32_f16(false, afr[1], false, pfrag[1],
                                                         (short)0, acc, false, false);
            #pragma unroll
            for (int v = 0; v < 8; ++v) {
                int m = v + hi * 8;
                out[((size_t)(bbase + m) * NCTX + s) * HID + np] = acc[v];
            }
        }

        // gates = bias + [h|x|0] @ B'  (12 WMMAs per wave per step)
        v8f g4[4];
        #pragma unroll
        for (int q = 0; q < 4; ++q) {
            v8f acc;
            #pragma unroll
            for (int v = 0; v < 8; ++v) acc[v] = gbias[q];
            acc = __builtin_amdgcn_wmma_f32_16x16x32_f16(false, afr[0], false, bfrag[q][0],
                                                         (short)0, acc, false, false);
            acc = __builtin_amdgcn_wmma_f32_16x16x32_f16(false, afr[1], false, bfrag[q][1],
                                                         (short)0, acc, false, false);
            acc = __builtin_amdgcn_wmma_f32_16x16x32_f16(false, afr[2], false, bfrag[q][2],
                                                         (short)0, acc, false, false);
            g4[q] = acc;
        }

        // elementwise LSTM cell update; i/f/g/o share (lane,v) -> no shuffles
        #pragma unroll
        for (int v = 0; v < 8; ++v) {
            float ig = fast_sigmoid(g4[0][v]);
            float fg = fast_sigmoid(g4[1][v]);
            float gg = fast_tanh(g4[2][v]);
            float og = fast_sigmoid(g4[3][v]);
            float c2 = fmaf(fg, cst[v], ig * gg);
            cst[v] = c2;
            float hv = og * fast_tanh(c2);
            int m = v + hi * 8;
            abuf[p ^ 1][m][np] = (_Float16)hv;   // h_t -> other buffer
        }
        p ^= 1;
    }

    // final context s=9 uses h_479 (now in abuf[p])
    __syncthreads();
    {
        union { v16h v; h8 h[2]; } u0, u1;
        const _Float16* b0 = &abuf[p][arow][koff];
        u0.h[0] = *(const h8*)(b0);
        u0.h[1] = *(const h8*)(b0 + 16);
        const _Float16* b1 = &abuf[p][arow][32 + koff];
        u1.h[0] = *(const h8*)(b1);
        u1.h[1] = *(const h8*)(b1 + 16);
        v8f acc;
        #pragma unroll
        for (int v = 0; v < 8; ++v) acc[v] = pbias;
        acc = __builtin_amdgcn_wmma_f32_16x16x32_f16(false, u0.v, false, pfrag[0],
                                                     (short)0, acc, false, false);
        acc = __builtin_amdgcn_wmma_f32_16x16x32_f16(false, u1.v, false, pfrag[1],
                                                     (short)0, acc, false, false);
        #pragma unroll
        for (int v = 0; v < 8; ++v) {
            int m = v + hi * 8;
            out[((size_t)(bbase + m) * NCTX + 9) * HID + np] = acc[v];
        }
    }
}

extern "C" void kernel_launch(void* const* d_in, const int* in_sizes, int n_in,
                              void* d_out, int out_size, void* d_ws, size_t ws_size,
                              hipStream_t stream) {
    (void)n_in; (void)out_size; (void)d_ws; (void)ws_size;
    const float* wi     = (const float*)d_in[0];
    const float* wfut   = (const float*)d_in[1];
    const float* W_ih   = (const float*)d_in[2];
    const float* W_hh   = (const float*)d_in[3];
    const float* b_ih   = (const float*)d_in[4];
    const float* b_hh   = (const float*)d_in[5];
    const float* W_proj = (const float*)d_in[6];
    const float* b_proj = (const float*)d_in[7];
    float* out = (float*)d_out;

    const int B = in_sizes[0] / (TIN * 3);   // 512
    const int nblocks = B / MROWS;           // 32 independent recurrences
    pwce_lstm_kernel<<<nblocks, THREADS, 0, stream>>>(
        wi, wfut, W_ih, W_hh, b_ih, b_hh, W_proj, b_proj, out);
}
